// VRNN_52089363366538
// MI455X (gfx1250) — matrix operand order repaired and checked by reading, not compiled, not run on hardware
//
#include <hip/hip_runtime.h>
#include <math.h>

#define B_SZ   64
#define T_SZ   256
#define XD     256
#define HD     1024
#define ZD     256
#define CATD   (XD + HD)     // 1280
#define G3     (3 * HD)      // 3072
#define DEC_CHUNK 2048
#define LOG2PI 1.8378770664093453f

typedef unsigned short u16;
typedef unsigned int   u32;
typedef unsigned long long u64;

typedef __bf16 bf16_t;
typedef bf16_t v16bf __attribute__((ext_vector_type(16)));
typedef float  v8f   __attribute__((ext_vector_type(8)));
typedef u32    v4u   __attribute__((ext_vector_type(4)));
typedef int    v4i   __attribute__((ext_vector_type(4)));
typedef int    v8i   __attribute__((ext_vector_type(8)));

// ---------------- small device helpers ----------------

__device__ __forceinline__ u16 f2bf(float f) {
    u32 u = __float_as_uint(f);
    u32 r = u + 0x7FFFu + ((u >> 16) & 1u);   // round-to-nearest-even
    return (u16)(r >> 16);
}

__device__ __forceinline__ float softplus_f(float x) {
    return (x > 20.0f) ? x : log1pf(expf(x));
}

__device__ __forceinline__ float sigmoid_f(float x) {
    return 1.0f / (1.0f + expf(-x));
}

// ---------------- Tensor Data Mover tile load ----------------
// DMA a 64(rows) x 32(cols) bf16 tile from a row-major [*, row_pitch] tensor
// into LDS (tile rows stored contiguously: 32 elems = 64B per row).
// D# packing per CDNA5 ISA 8.3/8.4:
//  group0: w0=count(1), w1=lds_addr, w2..w3 = global_addr[56:0] | type=2<<30
//  group1: data_size=1(2B), tensor_dim0=row_pitch, tensor_dim1=1<<20,
//          tile_dim0=32, tile_dim1=64, tensor_dim0_stride=row_pitch
__device__ __forceinline__ void tdm_load_tile(const u16* gbase, long elem_off,
                                              u32 lds_byte, int row_pitch) {
    u64 ga = (u64)gbase + 2ull * (u64)elem_off;
    v4u g0;
    g0[0] = 1u;                                           // count=1, user mode
    g0[1] = lds_byte;                                     // LDS dest (bytes)
    g0[2] = (u32)ga;                                      // global_addr[31:0]
    g0[3] = (u32)((ga >> 32) & 0x01FFFFFFu) | 0x80000000u; // addr[56:32] | type=2

    const u32 dim0 = (u32)row_pitch;        // tensor_dim0 (elements)
    const u32 dim1 = (1u << 20);            // tensor_dim1 (large, in-bounds tiles)
    const u64 str0 = (u64)row_pitch;        // tensor_dim0_stride (elements)
    v8i g1;
    g1[0] = (int)(1u << 16);                              // data_size=2 bytes
    g1[1] = (int)((dim0 & 0xFFFFu) << 16);                // dim0 lo16 -> [31:16]
    g1[2] = (int)(((dim0 >> 16) & 0xFFFFu) | ((dim1 & 0xFFFFu) << 16));
    g1[3] = (int)(((dim1 >> 16) & 0xFFFFu) | (32u << 16)); // tile_dim0=32
    g1[4] = (int)(64u);                                    // tile_dim1=64, tile_dim2=0
    g1[5] = (int)(u32)(str0 & 0xFFFFFFFFull);
    g1[6] = (int)(u32)((str0 >> 32) & 0xFFFFull);          // stride1=0
    g1[7] = 0;

    v4i z4 = {};
#if defined(__clang_major__) && (__clang_major__ >= 23)
    v8i z8 = {};
    __builtin_amdgcn_tensor_load_to_lds(g0, g1, z4, z4, z8, 0);
#else
    __builtin_amdgcn_tensor_load_to_lds(g0, g1, z4, z4, 0);
#endif
}

// ---------------- conversion kernels ----------------

// W [K,N] f32 row-major -> Wt [N,K] bf16 row-major (K contiguous)
__global__ void k_cvt_w_t(const float* __restrict__ W, u16* __restrict__ Wt,
                          int K, int N) {
    int idx = blockIdx.x * blockDim.x + threadIdx.x;
    long total = (long)K * N;
    if (idx >= total) return;
    int n = idx / K;
    int k = idx - n * K;
    Wt[idx] = f2bf(W[(long)k * N + n]);
}

__global__ void k_cvt(const float* __restrict__ in, u16* __restrict__ out, int n) {
    int i = blockIdx.x * blockDim.x + threadIdx.x;
    if (i < n) out[i] = f2bf(in[i]);
}

__global__ void k_zero32(u32* __restrict__ p, int n) {
    int i = blockIdx.x * blockDim.x + threadIdx.x;
    if (i < n) p[i] = 0u;
}

// ---------------- WMMA GEMM (TDM-fed, double-buffered) ----------------
// C[M,N] = act(A[M,K] @ W[K,N] + bias), A bf16 (row stride lda),
// W supplied transposed Bt[N,K] bf16 (row stride K).
// Block = 256 threads (8 waves, wave32). Block tile = 64(M) x 64(N).
// Wave 0 DMA-loads the A tile, wave 1 the B tile (TDM, TENSORcnt-tracked),
// double-buffered so tile i+1 streams in while tile i feeds the WMMAs.
// Fragment layouts used by this kernel (contiguous k, not the ISA table order):
//   A 16x32 bf16 : lanes 0-15 m=lane  K {0..15},        lanes 16-31 K {16..31}
//   B 32x16 bf16 : lanes 0-15 n=lane  K {0..15},        lanes 16-31 K {16..31}
//   C/D 16x16 f32: vgpr r -> m = r + 8*(lane>>4), n = lane&15

template <int ACT>  // 0=none 1=relu 2=softplus
__global__ __launch_bounds__(256) void k_gemm(
    const u16* __restrict__ A, int lda,
    const u16* __restrict__ Bt, int K,
    const float* __restrict__ bias,
    float* __restrict__ Cf, int ldcf,
    u16* __restrict__ Cb, int ldcb)
{
    __shared__ __align__(16) u16 As[2][64 * 32];
    __shared__ __align__(16) u16 Bs[2][64 * 32];

    const int tid  = threadIdx.x;
    const int lane = tid & 31;
    const int wave = tid >> 5;
    const int mt   = wave >> 1;
    const int nt0  = (wave & 1) * 2;

    const long m0 = (long)blockIdx.y * 64;
    const long n0 = (long)blockIdx.x * 64;

    const int lm = lane & 15;
    const int kh = lane >> 4;

    const u32 asBase = (u32)(size_t)(void*)&As[0][0];
    const u32 bsBase = (u32)(size_t)(void*)&Bs[0][0];

    v8f acc0 = {};
    v8f acc1 = {};

    const int nk = K >> 5;

    // prologue: DMA tile 0 into buffer 0
    if (wave == 0)      tdm_load_tile(A,  m0 * (long)lda, asBase, lda);
    else if (wave == 1) tdm_load_tile(Bt, n0 * (long)K,   bsBase, K);

    for (int i = 0; i < nk; ++i) {
        const int cur = i & 1;
        if (i + 1 < nk) {
            const long kn = (long)(i + 1) << 5;
            const u32 boff = (u32)((cur ^ 1) * 64 * 32 * 2);
            if (wave == 0)      tdm_load_tile(A,  m0 * (long)lda + kn, asBase + boff, lda);
            else if (wave == 1) tdm_load_tile(Bt, n0 * (long)K   + kn, bsBase + boff, K);
            if (wave < 2) __builtin_amdgcn_s_wait_tensorcnt(1);
        } else {
            if (wave < 2) __builtin_amdgcn_s_wait_tensorcnt(0);
        }
        __syncthreads();   // publish DMA'd tile to all waves

        const u16* Asc = &As[cur][0];
        const u16* Bsc = &Bs[cur][0];

        union { v16bf v; u32 u[8]; } af, bq0, bq1;
        const int am = mt * 16 + lm;
#pragma unroll
        for (int v = 0; v < 4; ++v) {
            af.u[v]     = *(const u32*)&Asc[am * 32 + kh * 16 + 2 * v];
            af.u[v + 4] = *(const u32*)&Asc[am * 32 + kh * 16 + 8 + 2 * v];
        }
        const int bn0 = (nt0 + 0) * 16 + lm;
        const int bn1 = (nt0 + 1) * 16 + lm;
#pragma unroll
        for (int v = 0; v < 8; ++v) {
            bq0.u[v] = *(const u32*)&Bsc[bn0 * 32 + kh * 16 + 2 * v];
            bq1.u[v] = *(const u32*)&Bsc[bn1 * 32 + kh * 16 + 2 * v];
        }
        acc0 = __builtin_amdgcn_wmma_f32_16x16x32_bf16(
                   false, af.v, false, bq0.v, (short)0, acc0, false, false);
        acc1 = __builtin_amdgcn_wmma_f32_16x16x32_bf16(
                   false, af.v, false, bq1.v, (short)0, acc1, false, false);
        __syncthreads();   // all reads done before next DMA overwrites buffer
    }

    // epilogue: bias + activation, write f32 and/or bf16
#pragma unroll
    for (int r = 0; r < 8; ++r) {
        const long grow = m0 + mt * 16 + r + 8 * kh;
        {
            const long gcol = n0 + (nt0 + 0) * 16 + lm;
            float v = acc0[r] + bias[gcol];
            if (ACT == 1) v = fmaxf(v, 0.0f);
            if (ACT == 2) v = softplus_f(v);
            if (Cf) Cf[grow * (long)ldcf + gcol] = v;
            if (Cb) Cb[grow * (long)ldcb + gcol] = f2bf(v);
        }
        {
            const long gcol = n0 + (nt0 + 1) * 16 + lm;
            float v = acc1[r] + bias[gcol];
            if (ACT == 1) v = fmaxf(v, 0.0f);
            if (ACT == 2) v = softplus_f(v);
            if (Cf) Cf[grow * (long)ldcf + gcol] = v;
            if (Cb) Cb[grow * (long)ldcb + gcol] = f2bf(v);
        }
    }
}

// ---------------- step elementwise kernels ----------------

// build enccat = [phi_x[:,t,:], h] (bf16) and snapshot hs[:,t,:] = h (pre-update)
__global__ void k_pack_enc(const u16* __restrict__ px, int t,
                           const u16* __restrict__ hbf,
                           u16* __restrict__ enccat, u16* __restrict__ hs) {
    int idx = blockIdx.x * blockDim.x + threadIdx.x;
    if (idx < B_SZ * CATD) {
        int b = idx / CATD;
        int j = idx - b * CATD;
        enccat[idx] = (j < XD) ? px[((long)b * T_SZ + t) * XD + j]
                               : hbf[b * HD + (j - XD)];
    }
    if (idx < B_SZ * HD) {
        int b = idx / HD;
        int j = idx - b * HD;
        hs[((long)b * T_SZ + t) * HD + j] = hbf[idx];
    }
}

// build gxcat = [phi_x[:,t,:], phi_z(z)] (bf16), phi_z read from pzs slice
__global__ void k_pack_gx(const u16* __restrict__ px, const u16* __restrict__ pzs,
                          int t, u16* __restrict__ gxcat) {
    int idx = blockIdx.x * blockDim.x + threadIdx.x;
    if (idx >= B_SZ * CATD) return;
    int b = idx / CATD;
    int j = idx - b * CATD;
    gxcat[idx] = (j < XD) ? px[((long)b * T_SZ + t) * XD + j]
                          : pzs[((long)b * T_SZ + t) * HD + (j - XD)];
}

// z = enc_mu + enc_sd * eps  (-> bf16), and accumulate KL per batch row
__global__ void k_zkld(const float* __restrict__ emu, const float* __restrict__ esd,
                       const float* __restrict__ pmu, const float* __restrict__ psd,
                       const float* __restrict__ eps, int t,
                       u16* __restrict__ zbf, float* __restrict__ klacc) {
    int b = blockIdx.x;     // 64 blocks
    int i = threadIdx.x;    // 256 threads == ZD
    int idx = b * ZD + i;
    float mu = emu[idx], sd = esd[idx];
    float pm = pmu[idx], ps = psd[idx];
    float e  = eps[((long)b * T_SZ + t) * ZD + i];
    zbf[idx] = f2bf(mu + sd * e);
    float d   = mu - pm;
    float kld = logf(ps / sd) + (sd * sd + d * d) / (2.0f * ps * ps) - 0.5f;
    __shared__ float red[256];
    red[i] = kld;
    __syncthreads();
    for (int s = 128; s > 0; s >>= 1) {
        if (i < s) red[i] += red[i + s];
        __syncthreads();
    }
    if (i == 0) klacc[b] += red[0];
}

// GRU gate combine (torch GRUCell r,u,n order)
__global__ void k_gru(const float* __restrict__ gi, const float* __restrict__ gh,
                      float* __restrict__ hbuf, u16* __restrict__ hbf) {
    int idx = blockIdx.x * blockDim.x + threadIdx.x;
    if (idx >= B_SZ * HD) return;
    int b = idx / HD;
    int j = idx - b * HD;
    const float* gib = gi + (long)b * G3;
    const float* ghb = gh + (long)b * G3;
    float r = sigmoid_f(gib[j]          + ghb[j]);
    float u = sigmoid_f(gib[HD + j]     + ghb[HD + j]);
    float n = tanhf   (gib[2 * HD + j] + r * ghb[2 * HD + j]);
    float h  = hbuf[idx];
    float hn = (1.0f - u) * n + u * h;
    hbuf[idx] = hn;
    hbf[idx]  = f2bf(hn);
}

// ---------------- decoder kernels ----------------

// deccat[i,:] = [pzs[row], hs[row]] for row = r0 + i  (rows are (b*T+t))
__global__ void k_pack_dec(const u16* __restrict__ pzs, const u16* __restrict__ hs,
                           int r0, u16* __restrict__ deccat) {
    int idx = blockIdx.x * blockDim.x + threadIdx.x;
    if (idx >= DEC_CHUNK * 2 * HD) return;
    int i = idx / (2 * HD);
    int j = idx - i * 2 * HD;
    long r = (long)r0 + i;
    deccat[idx] = (j < HD) ? pzs[r * HD + j] : hs[r * HD + (j - HD)];
}

// masked diagonal-Gaussian log-prob, accumulated into lpacc[b]
__global__ void k_loss(const float* __restrict__ mu, const float* __restrict__ sd,
                       const float* __restrict__ x, const int* __restrict__ x_sl,
                       int r0, float* __restrict__ lpacc) {
    int i   = threadIdx.x;      // 256 == XD
    int row = blockIdx.x;       // DEC_CHUNK blocks
    long r  = (long)r0 + row;
    int b   = (int)(r / T_SZ);
    int t   = (int)(r - (long)b * T_SZ);
    float m = mu[(long)row * XD + i];
    float s = sd[(long)row * XD + i];
    float xv = x[r * XD + i];
    float d = (xv - m) / s;
    float term = -logf(s) - 0.5f * LOG2PI - 0.5f * d * d;
    __shared__ float red[256];
    red[i] = term;
    __syncthreads();
    for (int sft = 128; sft > 0; sft >>= 1) {
        if (i < sft) red[i] += red[i + sft];
        __syncthreads();
    }
    if (i == 0 && t < x_sl[b]) atomicAdd(&lpacc[b], red[0]);
}

__global__ void k_final(const float* __restrict__ lpacc, const float* __restrict__ klacc,
                        const int* __restrict__ x_sl, float* __restrict__ out) {
    __shared__ float s1[64];
    __shared__ float s2[64];
    int i = threadIdx.x;   // 64 threads
    s1[i] = lpacc[i] - klacc[i];
    s2[i] = (float)x_sl[i];
    __syncthreads();
    for (int s = 32; s > 0; s >>= 1) {
        if (i < s) { s1[i] += s1[i + s]; s2[i] += s2[i + s]; }
        __syncthreads();
    }
    if (i == 0) out[0] = -s1[0] / s2[0];
}

// ---------------- host launcher ----------------

extern "C" void kernel_launch(void* const* d_in, const int* in_sizes, int n_in,
                              void* d_out, int out_size, void* d_ws, size_t ws_size,
                              hipStream_t stream) {
    const float* x    = (const float*)d_in[0];
    const int*   x_sl = (const int*)  d_in[1];
    const float* eps  = (const float*)d_in[2];

    char* ws = (char*)d_ws;
    size_t off = 0;
    auto alloc = [&](size_t bytes) -> void* {
        off = (off + 255) & ~(size_t)255;
        void* p = ws + off;
        off += bytes;
        return p;
    };

    // ---- bf16 transposed weights ----
    auto cvtW = [&](int idx, int K, int N) -> u16* {
        u16* Wt = (u16*)alloc((size_t)K * N * sizeof(u16));
        long total = (long)K * N;
        k_cvt_w_t<<<(int)((total + 255) / 256), 256, 0, stream>>>(
            (const float*)d_in[idx], Wt, K, N);
        return Wt;
    };
    auto bptr = [&](int idx) -> const float* { return (const float*)d_in[idx]; };

    u16* Wpx = cvtW(3,  XD,   XD);   const float* bpx = bptr(4);
    u16* Wz0 = cvtW(5,  ZD,   HD);   const float* bz0 = bptr(6);
    u16* Wz1 = cvtW(7,  HD,   HD);   const float* bz1 = bptr(8);
    u16* Wz2 = cvtW(9,  HD,   HD);   const float* bz2 = bptr(10);
    u16* Wz3 = cvtW(11, HD,   HD);   const float* bz3 = bptr(12);
    u16* Wp0 = cvtW(13, HD,   HD);   const float* bp0 = bptr(14);
    u16* Wp1 = cvtW(15, HD,   HD);   const float* bp1 = bptr(16);
    u16* Wp2 = cvtW(17, HD,   HD);   const float* bp2 = bptr(18);
    u16* Wpm = cvtW(19, HD,   ZD);   const float* bpm = bptr(20);
    u16* Wps = cvtW(21, HD,   ZD);   const float* bps = bptr(22);
    u16* We0 = cvtW(23, CATD, HD);   const float* be0 = bptr(24);
    u16* We1 = cvtW(25, HD,   HD);   const float* be1 = bptr(26);
    u16* We2 = cvtW(27, HD,   HD);   const float* be2 = bptr(28);
    u16* Wem = cvtW(29, HD,   ZD);   const float* bem = bptr(30);
    u16* Wes = cvtW(31, HD,   ZD);   const float* bes = bptr(32);
    u16* Wih = cvtW(33, CATD, G3);
    u16* Whh = cvtW(34, HD,   G3);
    const float* bih = bptr(35);
    const float* bhh = bptr(36);
    u16* Wd0 = cvtW(37, 2 * HD, HD); const float* bd0 = bptr(38);
    u16* Wd1 = cvtW(39, HD,   HD);   const float* bd1 = bptr(40);
    u16* Wd2 = cvtW(41, HD,   HD);   const float* bd2 = bptr(42);
    u16* Wlm = cvtW(43, HD,   XD);   const float* blm = bptr(44);
    u16* Wls = cvtW(45, HD,   XD);   const float* bls = bptr(46);

    // ---- activation / temp buffers ----
    const long NXT = (long)B_SZ * T_SZ * XD;   // 4.19M
    const long NHT = (long)B_SZ * T_SZ * HD;   // 16.7M
    u16*   xbf    = (u16*)  alloc(NXT * sizeof(u16));
    u16*   pxbf   = (u16*)  alloc(NXT * sizeof(u16));
    u16*   hs     = (u16*)  alloc(NHT * sizeof(u16));
    u16*   pzs    = (u16*)  alloc(NHT * sizeof(u16));
    float* hbuf   = (float*)alloc((size_t)B_SZ * HD * sizeof(float));
    u16*   hbf    = (u16*)  alloc((size_t)B_SZ * HD * sizeof(u16));
    u16*   tA     = (u16*)  alloc((size_t)B_SZ * HD * sizeof(u16));
    u16*   tB     = (u16*)  alloc((size_t)B_SZ * HD * sizeof(u16));
    u16*   enccat = (u16*)  alloc((size_t)B_SZ * CATD * sizeof(u16));
    u16*   gxcat  = (u16*)  alloc((size_t)B_SZ * CATD * sizeof(u16));
    float* pmu    = (float*)alloc((size_t)B_SZ * ZD * sizeof(float));
    float* psd    = (float*)alloc((size_t)B_SZ * ZD * sizeof(float));
    float* emu    = (float*)alloc((size_t)B_SZ * ZD * sizeof(float));
    float* esd    = (float*)alloc((size_t)B_SZ * ZD * sizeof(float));
    u16*   zbf    = (u16*)  alloc((size_t)B_SZ * ZD * sizeof(u16));
    float* gi     = (float*)alloc((size_t)B_SZ * G3 * sizeof(float));
    float* gh     = (float*)alloc((size_t)B_SZ * G3 * sizeof(float));
    u16*   deccat = (u16*)  alloc((size_t)DEC_CHUNK * 2 * HD * sizeof(u16));
    u16*   d0b    = (u16*)  alloc((size_t)DEC_CHUNK * HD * sizeof(u16));
    u16*   d1b    = (u16*)  alloc((size_t)DEC_CHUNK * HD * sizeof(u16));
    float* dmu    = (float*)alloc((size_t)DEC_CHUNK * XD * sizeof(float));
    float* dsd    = (float*)alloc((size_t)DEC_CHUNK * XD * sizeof(float));
    float* klacc  = (float*)alloc(64 * sizeof(float));
    float* lpacc  = (float*)alloc(64 * sizeof(float));
    (void)in_sizes; (void)n_in; (void)out_size; (void)ws_size;

    // ---- init ----
    k_zero32<<<(B_SZ * HD + 255) / 256, 256, 0, stream>>>((u32*)hbuf, B_SZ * HD);
    k_zero32<<<(B_SZ * HD / 2 + 255) / 256, 256, 0, stream>>>((u32*)hbf, B_SZ * HD / 2);
    k_zero32<<<1, 64, 0, stream>>>((u32*)klacc, 64);
    k_zero32<<<1, 64, 0, stream>>>((u32*)lpacc, 64);
    k_cvt<<<(int)((NXT + 255) / 256), 256, 0, stream>>>(x, xbf, (int)NXT);

    auto gemm = [&](int act, const u16* A, int lda, const u16* Bt, const float* bias,
                    int M, int N, int K, float* Cf, int ldcf, u16* Cb, int ldcb) {
        dim3 g(N / 64, M / 64);
        if (act == 0)
            k_gemm<0><<<g, 256, 0, stream>>>(A, lda, Bt, K, bias, Cf, ldcf, Cb, ldcb);
        else if (act == 1)
            k_gemm<1><<<g, 256, 0, stream>>>(A, lda, Bt, K, bias, Cf, ldcf, Cb, ldcb);
        else
            k_gemm<2><<<g, 256, 0, stream>>>(A, lda, Bt, K, bias, Cf, ldcf, Cb, ldcb);
    };

    // ---- phi_x over all timesteps: [B*T, X] = relu(x @ Wpx + b) ----
    gemm(1, xbf, XD, Wpx, bpx, B_SZ * T_SZ, XD, XD, nullptr, 0, pxbf, XD);

    // ---- recurrence ----
    const int PACK_GRID = (B_SZ * CATD + 255) / 256;
    for (int t = 0; t < T_SZ; ++t) {
        // prior path: h -> mlp x3 -> mu, softplus(sd)
        gemm(1, hbf, HD, Wp0, bp0, B_SZ, HD, HD, nullptr, 0, tA, HD);
        gemm(1, tA,  HD, Wp1, bp1, B_SZ, HD, HD, nullptr, 0, tB, HD);
        gemm(1, tB,  HD, Wp2, bp2, B_SZ, HD, HD, nullptr, 0, tA, HD);
        gemm(0, tA,  HD, Wpm, bpm, B_SZ, ZD, HD, pmu, ZD, nullptr, 0);
        gemm(2, tA,  HD, Wps, bps, B_SZ, ZD, HD, psd, ZD, nullptr, 0);

        // encoder path: [phi_x_t, h] -> mlp x3 -> mu, softplus(sd)
        k_pack_enc<<<PACK_GRID, 256, 0, stream>>>(pxbf, t, hbf, enccat, hs);
        gemm(1, enccat, CATD, We0, be0, B_SZ, HD, CATD, nullptr, 0, tA, HD);
        gemm(1, tA, HD, We1, be1, B_SZ, HD, HD, nullptr, 0, tB, HD);
        gemm(1, tB, HD, We2, be2, B_SZ, HD, HD, nullptr, 0, tA, HD);
        gemm(0, tA, HD, Wem, bem, B_SZ, ZD, HD, emu, ZD, nullptr, 0);
        gemm(2, tA, HD, Wes, bes, B_SZ, ZD, HD, esd, ZD, nullptr, 0);

        // sample z, accumulate KL
        k_zkld<<<B_SZ, 256, 0, stream>>>(emu, esd, pmu, psd, eps, t, zbf, klacc);

        // phi_z: z -> mlp x4, last layer written straight into pzs[:,t,:]
        gemm(1, zbf, ZD, Wz0, bz0, B_SZ, HD, ZD, nullptr, 0, tA, HD);
        gemm(1, tA, HD, Wz1, bz1, B_SZ, HD, HD, nullptr, 0, tB, HD);
        gemm(1, tB, HD, Wz2, bz2, B_SZ, HD, HD, nullptr, 0, tA, HD);
        gemm(1, tA, HD, Wz3, bz3, B_SZ, HD, HD, nullptr, 0, pzs + (long)t * HD, T_SZ * HD);

        // GRU gates
        k_pack_gx<<<PACK_GRID, 256, 0, stream>>>(pxbf, pzs, t, gxcat);
        gemm(0, gxcat, CATD, Wih, bih, B_SZ, G3, CATD, gi, G3, nullptr, 0);
        gemm(0, hbf,   HD,   Whh, bhh, B_SZ, G3, HD,   gh, G3, nullptr, 0);
        k_gru<<<(B_SZ * HD + 255) / 256, 256, 0, stream>>>(gi, gh, hbuf, hbf);
    }

    // ---- decoder + likelihood, chunked over B*T rows ----
    const int NROWS = B_SZ * T_SZ;
    for (int r0 = 0; r0 < NROWS; r0 += DEC_CHUNK) {
        k_pack_dec<<<(DEC_CHUNK * 2 * HD + 255) / 256, 256, 0, stream>>>(
            pzs, hs, r0, deccat);
        gemm(1, deccat, 2 * HD, Wd0, bd0, DEC_CHUNK, HD, 2 * HD, nullptr, 0, d0b, HD);
        gemm(1, d0b, HD, Wd1, bd1, DEC_CHUNK, HD, HD, nullptr, 0, d1b, HD);
        gemm(1, d1b, HD, Wd2, bd2, DEC_CHUNK, HD, HD, nullptr, 0, d0b, HD);
        gemm(0, d0b, HD, Wlm, blm, DEC_CHUNK, XD, HD, dmu, XD, nullptr, 0);
        gemm(2, d0b, HD, Wls, bls, DEC_CHUNK, XD, HD, dsd, XD, nullptr, 0);
        k_loss<<<DEC_CHUNK, 256, 0, stream>>>(dmu, dsd, x, x_sl, r0, lpacc);
    }

    // ---- final scalar loss ----
    k_final<<<1, 64, 0, stream>>>(lpacc, klacc, x_sl, (float*)d_out);
}
